// Encoder_layer_71536975282556
// MI455X (gfx1250) — compile-verified, matrix operand order checked by reading, not run on hardware
//
#include <hip/hip_runtime.h>
#include <cstdint>
#include <cstddef>

#define D_MODEL 1024
#define N_HEADS 16
#define DKH     64
#define D_FF    4096
#define BATCH   2
#define SEQ     2048
#define M_ROWS  (BATCH * SEQ)   // 4096

typedef __attribute__((ext_vector_type(16))) __bf16 v16bf;
typedef __attribute__((ext_vector_type(8)))  float  v8f;

union AFrag { v16bf v; unsigned short u[16]; unsigned w[8]; };
union U4    { uint4 q; unsigned short s[8]; };

__device__ __forceinline__ unsigned short f32_to_bf16(float f) {
    union { float f; unsigned u; } x; x.f = f;
    unsigned r = x.u + 0x7FFFu + ((x.u >> 16) & 1u);
    return (unsigned short)(r >> 16);
}

// K-pair base index for 16-bit A-layout fragments (ISA 7.12.2):
// lanes 0-15: VGPR0..3 -> K=0..7, VGPR4..7 -> K=16..23 ; lanes 16-31: +8
__device__ __forceinline__ int kpair(int v, int half) {
    return (v < 4 ? 2 * v : 16 + 2 * (v - 4)) + 8 * half;
}

// ---------------------------------------------------------------------------
// fp32 -> bf16 conversion (grid-stride)
// ---------------------------------------------------------------------------
__global__ void cvt_bf16_kernel(const float* __restrict__ in,
                                unsigned short* __restrict__ out, size_t n) {
    size_t i = (size_t)blockIdx.x * blockDim.x + threadIdx.x;
    size_t stride = (size_t)gridDim.x * blockDim.x;
    for (; i < n; i += stride) out[i] = f32_to_bf16(in[i]);
}

// ---------------------------------------------------------------------------
// bf16 WMMA GEMM: C[M,N] = A[M,K] @ B[K,N] + bias, optional ReLU.
// Block: 256 threads (8 waves), tile 64(M) x 128(N), K-chunk 32.
// B tile is stored TRANSPOSED in LDS so fragment K-pairs are contiguous.
// mode: 0 = f32 row-major, 1 = bf16 row-major, 2 = bf16 head-split [B,H,S,Dk]
// ---------------------------------------------------------------------------
__global__ __launch_bounds__(256) void gemm_bf16_kernel(
    const unsigned short* __restrict__ A, const unsigned short* __restrict__ B,
    const float* __restrict__ bias, void* __restrict__ Cout,
    int M, int N, int K, int mode, int relu) {
    __shared__ __align__(16) unsigned short As[64][48];   // [m][k], 96B rows
    __shared__ __align__(16) unsigned short BsT[128][34]; // [n][k]

    const int t     = threadIdx.x;
    const int mBase = blockIdx.y * 64;
    const int nBase = blockIdx.x * 128;
    const int wave  = t >> 5, lane = t & 31;
    const int wm    = wave >> 2, wn = wave & 3;
    const int half  = lane >> 4, ln = lane & 15;

    const v8f zero = {0.f, 0.f, 0.f, 0.f, 0.f, 0.f, 0.f, 0.f};
    v8f acc[2][2];
    acc[0][0] = zero; acc[0][1] = zero; acc[1][0] = zero; acc[1][1] = zero;

    const int arow = t >> 2, acol = (t & 3) * 8;   // A: 64 rows x 32 cols
    const int brow = t >> 3, bcol = (t & 7) * 16;  // B: 32 rows x 128 cols

    for (int kc = 0; kc < K; kc += 32) {
        U4 av;
        av.q = *reinterpret_cast<const uint4*>(
            A + (size_t)(mBase + arow) * K + kc + acol);
        *reinterpret_cast<uint4*>(&As[arow][acol]) = av.q;

        const unsigned short* bp = B + (size_t)(kc + brow) * N + nBase + bcol;
        U4 bv0, bv1;
        bv0.q = *reinterpret_cast<const uint4*>(bp);
        bv1.q = *reinterpret_cast<const uint4*>(bp + 8);
#pragma unroll
        for (int e = 0; e < 8; ++e) {
            BsT[bcol + e][brow]     = bv0.s[e];
            BsT[bcol + 8 + e][brow] = bv1.s[e];
        }
        if (kc + 32 < K) {
            __builtin_prefetch(A + (size_t)(mBase + arow) * K + kc + 32 + acol, 0, 1);
            __builtin_prefetch(B + (size_t)(kc + 32 + brow) * N + nBase + bcol, 0, 1);
        }
        __syncthreads();

        AFrag a[2], b[2];
#pragma unroll
        for (int i = 0; i < 2; ++i)
#pragma unroll
            for (int v = 0; v < 8; ++v) {
                int k = kpair(v, half);
                int m = wm * 32 + i * 16 + ln;
                a[i].w[v] = *reinterpret_cast<const unsigned*>(&As[m][k]);
            }
#pragma unroll
        for (int j = 0; j < 2; ++j)
#pragma unroll
            for (int v = 0; v < 8; ++v) {
                int k = kpair(v, half);
                int n = wn * 32 + j * 16 + ln;
                b[j].w[v] = *reinterpret_cast<const unsigned*>(&BsT[n][k]);
            }
#pragma unroll
        for (int i = 0; i < 2; ++i)
#pragma unroll
            for (int j = 0; j < 2; ++j)
                acc[i][j] = __builtin_amdgcn_wmma_f32_16x16x32_bf16(
                    false, a[i].v, false, b[j].v, (short)0, acc[i][j],
                    false, false);
        __syncthreads();
    }

#pragma unroll
    for (int i = 0; i < 2; ++i)
#pragma unroll
        for (int j = 0; j < 2; ++j)
#pragma unroll
            for (int r = 0; r < 8; ++r) {
                int row = mBase + wm * 32 + i * 16 + r + 8 * half;
                int col = nBase + wn * 32 + j * 16 + ln;
                float val = acc[i][j][r] + bias[col];
                if (relu) val = fmaxf(val, 0.f);
                if (mode == 0) {
                    ((float*)Cout)[(size_t)row * N + col] = val;
                } else if (mode == 1) {
                    ((unsigned short*)Cout)[(size_t)row * N + col] =
                        f32_to_bf16(val);
                } else {
                    int bb = row / SEQ, s = row % SEQ;
                    int hh = col / DKH, d = col % DKH;
                    ((unsigned short*)Cout)[(((size_t)(bb * N_HEADS + hh)) * SEQ + s) * DKH + d] =
                        f32_to_bf16(val);
                }
            }
}

// ---------------------------------------------------------------------------
// Flash attention: grid(S/64, B*H), block 128 (4 waves, 16 q-rows each).
// K and V^T tiles staged in LDS once per 32-key block (shared by all waves).
// Online softmax; 8 WMMAs per key tile per wave.
// ---------------------------------------------------------------------------
__global__ __launch_bounds__(128) void attn_kernel(
    const unsigned short* __restrict__ Q, const unsigned short* __restrict__ Kt,
    const unsigned short* __restrict__ V, const int* __restrict__ mask,
    unsigned short* __restrict__ ctx) {
    __shared__ __align__(16) unsigned short Ks[32][72];   // [key][dk], 144B rows
    __shared__ __align__(16) unsigned short Vt[64][34];   // [dk][key]
    __shared__ __align__(16) unsigned short Plds[4][16][34];

    const int t    = threadIdx.x;
    const int w    = t >> 5, lane = t & 31;
    const int half = lane >> 4, ln = lane & 15;
    const int bh   = blockIdx.y, b = bh >> 4, h = bh & 15;
    const size_t headBase = (size_t)bh * SEQ * DKH;
    const int qbase = blockIdx.x * 64 + w * 16;
    const float NEG_INF = -__builtin_inff();

    // Q fragments for two 32-wide K-chunks of Dk=64 (contiguous dword pairs)
    AFrag qf[2];
#pragma unroll
    for (int c = 0; c < 2; ++c)
#pragma unroll
        for (int v = 0; v < 8; ++v) {
            int k = kpair(v, half) + 32 * c;
            qf[c].w[v] = *reinterpret_cast<const unsigned*>(
                Q + headBase + (size_t)(qbase + ln) * DKH + k);
        }

    const v8f zero = {0.f, 0.f, 0.f, 0.f, 0.f, 0.f, 0.f, 0.f};
    v8f acc[4] = {zero, zero, zero, zero};
    float mrow[8], lrow[8];
#pragma unroll
    for (int r = 0; r < 8; ++r) { mrow[r] = NEG_INF; lrow[r] = 0.f; }

    const int skey = t >> 2;          // 0..31 (staging key)
    const int sdk  = (t & 3) * 16;    // 0,16,32,48 (staging dk chunk)

    for (int kt = 0; kt < SEQ / 32; ++kt) {
        const int keyb = kt * 32;
        __syncthreads();  // prior iteration done reading Ks/Vt/Plds

        // cooperative staging: K tile [32][64] and V^T tile [64][32]
        {
            const unsigned short* kp =
                Kt + headBase + (size_t)(keyb + skey) * DKH + sdk;
            U4 k0, k1;
            k0.q = *reinterpret_cast<const uint4*>(kp);
            k1.q = *reinterpret_cast<const uint4*>(kp + 8);
            *reinterpret_cast<uint4*>(&Ks[skey][sdk])     = k0.q;
            *reinterpret_cast<uint4*>(&Ks[skey][sdk + 8]) = k1.q;

            const unsigned short* vp =
                V + headBase + (size_t)(keyb + skey) * DKH + sdk;
            U4 v0, v1;
            v0.q = *reinterpret_cast<const uint4*>(vp);
            v1.q = *reinterpret_cast<const uint4*>(vp + 8);
#pragma unroll
            for (int e = 0; e < 8; ++e) {
                Vt[sdk + e][skey]     = v0.s[e];
                Vt[sdk + 8 + e][skey] = v1.s[e];
            }
        }
        __syncthreads();

        v8f s[2] = {zero, zero};
#pragma unroll
        for (int c = 0; c < 2; ++c)
#pragma unroll
            for (int tt = 0; tt < 2; ++tt) {
                AFrag kf;
#pragma unroll
                for (int v = 0; v < 8; ++v) {
                    int k = kpair(v, half) + 32 * c;
                    kf.w[v] = *reinterpret_cast<const unsigned*>(
                        &Ks[tt * 16 + ln][k]);
                }
                s[tt] = __builtin_amdgcn_wmma_f32_16x16x32_bf16(
                    false, qf[c].v, false, kf.v, (short)0, s[tt], false, false);
            }

        const int msk0 = mask[b * SEQ + keyb + ln];
        const int msk1 = mask[b * SEQ + keyb + 16 + ln];
#pragma unroll
        for (int r = 0; r < 8; ++r) {
            s[0][r] = msk0 ? s[0][r] * 0.125f : NEG_INF;
            s[1][r] = msk1 ? s[1][r] * 0.125f : NEG_INF;
        }
        // online softmax (reductions stay inside each 16-lane half-group)
#pragma unroll
        for (int r = 0; r < 8; ++r) {
            float rv = fmaxf(s[0][r], s[1][r]);
            rv = fmaxf(rv, __shfl_xor(rv, 1, 32));
            rv = fmaxf(rv, __shfl_xor(rv, 2, 32));
            rv = fmaxf(rv, __shfl_xor(rv, 4, 32));
            rv = fmaxf(rv, __shfl_xor(rv, 8, 32));
            float mo = mrow[r];
            float mn = fmaxf(mo, rv);
            float alpha = (mo == NEG_INF) ? 0.f : __expf(mo - mn);
            float p0 = (mn == NEG_INF) ? 0.f : __expf(s[0][r] - mn);
            float p1 = (mn == NEG_INF) ? 0.f : __expf(s[1][r] - mn);
            s[0][r] = p0; s[1][r] = p1;
            float rs = p0 + p1;
            rs += __shfl_xor(rs, 1, 32);
            rs += __shfl_xor(rs, 2, 32);
            rs += __shfl_xor(rs, 4, 32);
            rs += __shfl_xor(rs, 8, 32);
            lrow[r] = lrow[r] * alpha + rs;
            mrow[r] = mn;
            acc[0][r] *= alpha; acc[1][r] *= alpha;
            acc[2][r] *= alpha; acc[3][r] *= alpha;
        }
        // bounce P through LDS to convert C-layout -> A-layout
#pragma unroll
        for (int tt = 0; tt < 2; ++tt)
#pragma unroll
            for (int r = 0; r < 8; ++r)
                Plds[w][r + 8 * half][tt * 16 + ln] = f32_to_bf16(s[tt][r]);
        __syncthreads();

        AFrag pf;
#pragma unroll
        for (int v = 0; v < 8; ++v) {
            int k = kpair(v, half);
            pf.w[v] = *reinterpret_cast<const unsigned*>(&Plds[w][ln][k]);
        }
#pragma unroll
        for (int j = 0; j < 4; ++j) {
            AFrag vf;
#pragma unroll
            for (int v = 0; v < 8; ++v) {
                int k = kpair(v, half);
                vf.w[v] = *reinterpret_cast<const unsigned*>(
                    &Vt[j * 16 + ln][k]);
            }
            acc[j] = __builtin_amdgcn_wmma_f32_16x16x32_bf16(
                false, pf.v, false, vf.v, (short)0, acc[j], false, false);
        }
    }

#pragma unroll
    for (int j = 0; j < 4; ++j)
#pragma unroll
        for (int r = 0; r < 8; ++r) {
            float denom = lrow[r];
            float val = denom > 0.f ? acc[j][r] / denom : 0.f;
            int row = qbase + r + 8 * half;
            size_t dst = ((size_t)(b * SEQ + row)) * D_MODEL + h * DKH + j * 16 + ln;
            ctx[dst] = f32_to_bf16(val);
        }
}

// ---------------------------------------------------------------------------
// out = LayerNorm(x + y) ; optionally also emit bf16 copy. One block per row.
// ---------------------------------------------------------------------------
__global__ __launch_bounds__(256) void add_ln_kernel(
    const float* __restrict__ x, const float* __restrict__ y,
    const float* __restrict__ g, const float* __restrict__ beta,
    float* __restrict__ outf, unsigned short* __restrict__ outb) {
    __shared__ float red[256];
    __shared__ float buf[D_MODEL];
    const int row = blockIdx.x, t = threadIdx.x;
    const size_t base = (size_t)row * D_MODEL;

    float s = 0.f;
    for (int d = t; d < D_MODEL; d += 256) {
        float v = x[base + d] + y[base + d];
        buf[d] = v;
        s += v;
    }
    red[t] = s;
    __syncthreads();
    for (int o = 128; o > 0; o >>= 1) {
        if (t < o) red[t] += red[t + o];
        __syncthreads();
    }
    const float mu = red[0] * (1.f / D_MODEL);
    __syncthreads();

    float vs = 0.f;
    for (int d = t; d < D_MODEL; d += 256) {
        float dd = buf[d] - mu;
        vs += dd * dd;
    }
    red[t] = vs;
    __syncthreads();
    for (int o = 128; o > 0; o >>= 1) {
        if (t < o) red[t] += red[t + o];
        __syncthreads();
    }
    const float inv = rsqrtf(red[0] * (1.f / D_MODEL) + 1e-5f);

    for (int d = t; d < D_MODEL; d += 256) {
        float o = (buf[d] - mu) * inv * g[d] + beta[d];
        outf[base + d] = o;
        if (outb) outb[base + d] = f32_to_bf16(o);
    }
}

// ---------------------------------------------------------------------------
extern "C" void kernel_launch(void* const* d_in, const int* in_sizes, int n_in,
                              void* d_out, int out_size, void* d_ws, size_t ws_size,
                              hipStream_t stream) {
    const float* x    = (const float*)d_in[0];
    const int*   mask = (const int*)d_in[1];
    const float* Wq = (const float*)d_in[2];  const float* bq = (const float*)d_in[3];
    const float* Wk = (const float*)d_in[4];  const float* bk = (const float*)d_in[5];
    const float* Wv = (const float*)d_in[6];  const float* bv = (const float*)d_in[7];
    const float* Wo = (const float*)d_in[8];  const float* bo = (const float*)d_in[9];
    const float* W1 = (const float*)d_in[10]; const float* b1 = (const float*)d_in[11];
    const float* W2 = (const float*)d_in[12]; const float* b2 = (const float*)d_in[13];
    const float* g1 = (const float*)d_in[14]; const float* be1 = (const float*)d_in[15];
    const float* g2 = (const float*)d_in[16]; const float* be2 = (const float*)d_in[17];

    char* ws = (char*)d_ws;
    size_t off = 0;
    auto alloc = [&](size_t bytes) -> void* {
        void* p = ws + off;
        off += (bytes + 255) & ~(size_t)255;
        return p;
    };

    unsigned short* xb  = (unsigned short*)alloc((size_t)M_ROWS * D_MODEL * 2);
    unsigned short* Wqb = (unsigned short*)alloc((size_t)D_MODEL * D_MODEL * 2);
    unsigned short* Wkb = (unsigned short*)alloc((size_t)D_MODEL * D_MODEL * 2);
    unsigned short* Wvb = (unsigned short*)alloc((size_t)D_MODEL * D_MODEL * 2);
    unsigned short* Wob = (unsigned short*)alloc((size_t)D_MODEL * D_MODEL * 2);
    unsigned short* W1b = (unsigned short*)alloc((size_t)D_MODEL * D_FF * 2);
    unsigned short* W2b = (unsigned short*)alloc((size_t)D_FF * D_MODEL * 2);
    unsigned short* Qb  = (unsigned short*)alloc((size_t)M_ROWS * D_MODEL * 2);
    unsigned short* Kb  = (unsigned short*)alloc((size_t)M_ROWS * D_MODEL * 2);
    unsigned short* Vb  = (unsigned short*)alloc((size_t)M_ROWS * D_MODEL * 2);
    unsigned short* ctxb = (unsigned short*)alloc((size_t)M_ROWS * D_MODEL * 2);
    float* attn_out = (float*)alloc((size_t)M_ROWS * D_MODEL * 4);
    float* hbuf     = (float*)alloc((size_t)M_ROWS * D_MODEL * 4);
    unsigned short* hb = (unsigned short*)alloc((size_t)M_ROWS * D_MODEL * 2);
    // reuse dead regions: Q/K/V/ctx (4 x 8MB contiguous) -> FF1 activations (32MB)
    unsigned short* ff1b = Qb;
    float* ff2 = attn_out;

    // fp32 -> bf16 conversions
    cvt_bf16_kernel<<<2048, 256, 0, stream>>>(x,  xb,  (size_t)M_ROWS * D_MODEL);
    cvt_bf16_kernel<<<1024, 256, 0, stream>>>(Wq, Wqb, (size_t)D_MODEL * D_MODEL);
    cvt_bf16_kernel<<<1024, 256, 0, stream>>>(Wk, Wkb, (size_t)D_MODEL * D_MODEL);
    cvt_bf16_kernel<<<1024, 256, 0, stream>>>(Wv, Wvb, (size_t)D_MODEL * D_MODEL);
    cvt_bf16_kernel<<<1024, 256, 0, stream>>>(Wo, Wob, (size_t)D_MODEL * D_MODEL);
    cvt_bf16_kernel<<<2048, 256, 0, stream>>>(W1, W1b, (size_t)D_MODEL * D_FF);
    cvt_bf16_kernel<<<2048, 256, 0, stream>>>(W2, W2b, (size_t)D_FF * D_MODEL);

    // QKV projections (bf16 WMMA, head-split bf16 outputs)
    dim3 gQ(D_MODEL / 128, M_ROWS / 64);
    gemm_bf16_kernel<<<gQ, 256, 0, stream>>>(xb, Wqb, bq, Qb, M_ROWS, D_MODEL, D_MODEL, 2, 0);
    gemm_bf16_kernel<<<gQ, 256, 0, stream>>>(xb, Wkb, bk, Kb, M_ROWS, D_MODEL, D_MODEL, 2, 0);
    gemm_bf16_kernel<<<gQ, 256, 0, stream>>>(xb, Wvb, bv, Vb, M_ROWS, D_MODEL, D_MODEL, 2, 0);

    // flash attention -> ctx (bf16, merged-head layout)
    attn_kernel<<<dim3(SEQ / 64, BATCH * N_HEADS), 128, 0, stream>>>(Qb, Kb, Vb, mask, ctxb);

    // output projection -> f32
    gemm_bf16_kernel<<<gQ, 256, 0, stream>>>(ctxb, Wob, bo, attn_out, M_ROWS, D_MODEL, D_MODEL, 0, 0);

    // h = LN(x + attn_out), emit f32 + bf16
    add_ln_kernel<<<M_ROWS, 256, 0, stream>>>(x, attn_out, g1, be1, hbuf, hb);

    // FFN: relu(h @ W1 + b1) -> bf16 ; @ W2 + b2 -> f32
    dim3 gF1(D_FF / 128, M_ROWS / 64);
    gemm_bf16_kernel<<<gF1, 256, 0, stream>>>(hb, W1b, b1, ff1b, M_ROWS, D_FF, D_MODEL, 1, 1);
    dim3 gF2(D_MODEL / 128, M_ROWS / 64);
    gemm_bf16_kernel<<<gF2, 256, 0, stream>>>(ff1b, W2b, b2, ff2, M_ROWS, D_MODEL, D_FF, 0, 0);

    // out = LN(h + ff)
    add_ln_kernel<<<M_ROWS, 256, 0, stream>>>(hbuf, ff2, g2, be2, (float*)d_out, nullptr);
    (void)in_sizes; (void)n_in; (void)out_size; (void)ws_size;
}
